// LSTMAutoencoder_21852793602518
// MI455X (gfx1250) — matrix operand order repaired and checked by reading, not compiled, not run on hardware
//
#include <hip/hip_runtime.h>

#define BATCH 32768
#define SEQ   10
#define DIN   46
#define HID   64
#define NG    256   // 4*HID
#define KP    64    // K padded to 64
#define WPN   48    // projection N padded to 48

typedef __attribute__((ext_vector_type(16))) _Float16 v16h;
typedef __attribute__((ext_vector_type(8)))  float    v8f;

union AF { v16h v; _Float16 h[16]; unsigned u[8]; };

__device__ __forceinline__ int kbase(int v, bool hi) {
  // 16-bit A-matrix 16x32 layout: lanes 0-15 K={0..7,16..23}, lanes 16-31 K={8..15,24..31}
  return (v < 4 ? 2 * v : 16 + 2 * (v - 4)) + (hi ? 8 : 0);
}

__device__ __forceinline__ AF load_afrag_lds(const _Float16* rowbase, int ks, bool hi) {
  AF a;
#pragma unroll
  for (int v = 0; v < 8; ++v) {
    int K = kbase(v, hi) + 32 * ks;
    a.u[v] = *reinterpret_cast<const unsigned*>(rowbase + K);
  }
  return a;
}

__device__ __forceinline__ AF load_bfrag(const _Float16* W, int stride, int ntile, int ks, int lane) {
  // B 32x16: lane%16 = N column, elements over K in same order as A (kbase)
  AF b;
  const _Float16* p = W + (size_t)(32 * ks) * stride + ntile * 16 + (lane & 15);
#pragma unroll
  for (int v = 0; v < 8; ++v) {
    int K = kbase(v, lane >= 16);
    b.h[2 * v]     = p[(size_t)K * stride];
    b.h[2 * v + 1] = p[(size_t)(K + 1) * stride];
  }
  return b;
}

__device__ __forceinline__ float fast_rcp(float x) {
#if __has_builtin(__builtin_amdgcn_rcpf)
  return __builtin_amdgcn_rcpf(x);
#else
  return 1.0f / x;
#endif
}
__device__ __forceinline__ float fast_sigmoid(float x) {
  return fast_rcp(1.0f + __expf(-x));
}
__device__ __forceinline__ float fast_tanh(float x) {
#if __has_builtin(__builtin_amdgcn_tanhf)
  return __builtin_amdgcn_tanhf(x);   // gfx1250 v_tanh_f32
#else
  return 1.0f - 2.0f * fast_rcp(__expf(2.0f * x) + 1.0f);
#endif
}

#define WMMA_F16(A, B, C) \
  __builtin_amdgcn_wmma_f32_16x16x32_f16(false, (A), false, (B), (short)0, (C), false, false)

__global__ __launch_bounds__(256, 1)
void lstm_ae_kernel(const float* __restrict__ x,
                    const float* __restrict__ Wih_e, const float* __restrict__ Whh_e,
                    const float* __restrict__ bih_e, const float* __restrict__ bhh_e,
                    const float* __restrict__ Wih_d, const float* __restrict__ Whh_d,
                    const float* __restrict__ bih_d, const float* __restrict__ bhh_d,
                    const float* __restrict__ Wp,    const float* __restrict__ bp,
                    float* __restrict__ out)
{
  __shared__ __align__(16) _Float16 sWx[KP * NG];   // encoder Wih^T, K padded (32 KB)
  __shared__ __align__(16) _Float16 sWh[KP * NG];   // encoder Whh^T          (32 KB)
  __shared__ __align__(16) _Float16 sWd[KP * NG];   // (Wih_d + Whh_d)^T      (32 KB)
  __shared__ __align__(16) _Float16 sWp[KP * WPN];  // Wp^T, N padded         ( 6 KB)
  __shared__ float sBe[NG], sBd[NG], sBp[WPN];
  __shared__ __align__(16) _Float16 sH[8 * 16 * HID];  // per-wave h staging (16 KB)

  const int tid = threadIdx.x;

  // ---- cooperative weight load: global f32 -> LDS f16, transposed to [K][N] ----
  for (int idx = tid; idx < KP * NG; idx += 256) {
    int k = idx >> 8, n = idx & 255;
    sWx[idx] = (k < DIN) ? (_Float16)Wih_e[n * DIN + k] : (_Float16)0.f;
    sWh[idx] = (_Float16)Whh_e[n * HID + k];
    sWd[idx] = (_Float16)(Wih_d[n * HID + k] + Whh_d[n * HID + k]);
  }
  for (int idx = tid; idx < KP * WPN; idx += 256) {
    int k = idx / WPN, n = idx % WPN;
    sWp[idx] = (n < DIN) ? (_Float16)Wp[n * HID + k] : (_Float16)0.f;
  }
  if (tid < NG) {
    sBe[tid] = bih_e[tid] + bhh_e[tid];
    sBd[tid] = bih_d[tid] + bhh_d[tid];
  }
  if (tid < WPN) sBp[tid] = (tid < DIN) ? bp[tid] : 0.f;
  for (int idx = tid; idx < 8 * 16 * HID; idx += 256) sH[idx] = (_Float16)0.f;
  __syncthreads();

  const int  lane   = tid & 31;
  const int  wave   = tid >> 5;
  const bool hi     = lane >= 16;
  const int  lanelo = lane & 15;
  const int  mBase  = blockIdx.x * 128 + wave * 16;

  _Float16* hstage = sH + wave * 16 * HID;           // this wave's 16x64 h tile
  const _Float16* harow = hstage + lanelo * HID;     // A-fragment row for this lane

  float c[4][8];
#pragma unroll
  for (int q = 0; q < 4; ++q)
#pragma unroll
    for (int j = 0; j < 8; ++j) c[q][j] = 0.f;

  const float* xrow = x + (size_t)(mBase + lanelo) * SEQ * DIN;
  float* enc_out = out + (size_t)BATCH * SEQ * DIN;

  // =========================== encoder ===========================
#pragma unroll 1
  for (int t = 0; t < SEQ; ++t) {
    // Defeat LICM: weights in LDS must be re-read each step (keeps them as
    // ds_loads near their WMMA instead of hoisted + spilled to scratch).
    asm volatile("" ::: "memory");

    AF ax[2];
#pragma unroll
    for (int ks = 0; ks < 2; ++ks) {
#pragma unroll
      for (int v = 0; v < 8; ++v) {
        int K = kbase(v, hi) + 32 * ks;
        float2 val = make_float2(0.f, 0.f);
        if (K < DIN)  // K even, so K+1 <= 45 is also valid
          val = *reinterpret_cast<const float2*>(xrow + t * DIN + K);
        ax[ks].h[2 * v]     = (_Float16)val.x;
        ax[ks].h[2 * v + 1] = (_Float16)val.y;
      }
    }
    AF ah0 = load_afrag_lds(harow, 0, hi);
    AF ah1 = load_afrag_lds(harow, 1, hi);

#pragma unroll
    for (int ht = 0; ht < 4; ++ht) {
      v8f acc[4];
#pragma unroll
      for (int g = 0; g < 4; ++g) {
        int nt = 4 * g + ht;
        v8f a = {};   // inline-0 accumulator; bias folded into activation
        AF b0 = load_bfrag(sWx, NG, nt, 0, lane);
        a = WMMA_F16(ax[0].v, b0.v, a);
        AF b1 = load_bfrag(sWx, NG, nt, 1, lane);
        a = WMMA_F16(ax[1].v, b1.v, a);
        AF b2 = load_bfrag(sWh, NG, nt, 0, lane);
        a = WMMA_F16(ah0.v, b2.v, a);
        AF b3 = load_bfrag(sWh, NG, nt, 1, lane);
        a = WMMA_F16(ah1.v, b3.v, a);
        acc[g] = a;
      }
      float bi = sBe[16 * (0 * 4 + ht) + lanelo];
      float bf = sBe[16 * (1 * 4 + ht) + lanelo];
      float bg = sBe[16 * (2 * 4 + ht) + lanelo];
      float bo = sBe[16 * (3 * 4 + ht) + lanelo];
#pragma unroll
      for (int j = 0; j < 8; ++j) {
        float ig = fast_sigmoid(acc[0][j] + bi);
        float fg = fast_sigmoid(acc[1][j] + bf);
        float gg = fast_tanh(acc[2][j] + bg);
        float og = fast_sigmoid(acc[3][j] + bo);
        float cn = fg * c[ht][j] + ig * gg;
        c[ht][j] = cn;
        float hn = og * fast_tanh(cn);
        int m = j + (hi ? 8 : 0);
        hstage[m * HID + 16 * ht + lanelo] = (_Float16)hn;
        if (t == SEQ - 1)
          enc_out[(size_t)(mBase + m) * HID + 16 * ht + lanelo] = hn;
      }
    }
  }

  // =========================== decoder ===========================
  // Reference: carry=(h,h,c) -> decoder input == hidden state every step,
  // so gates = h @ (Wih_d + Whh_d)^T + bd exactly.
#pragma unroll 1
  for (int t = 0; t < SEQ; ++t) {
    asm volatile("" ::: "memory");

    AF ah0 = load_afrag_lds(harow, 0, hi);
    AF ah1 = load_afrag_lds(harow, 1, hi);

#pragma unroll
    for (int ht = 0; ht < 4; ++ht) {
      v8f acc[4];
#pragma unroll
      for (int g = 0; g < 4; ++g) {
        int nt = 4 * g + ht;
        v8f a = {};
        AF b0 = load_bfrag(sWd, NG, nt, 0, lane);
        a = WMMA_F16(ah0.v, b0.v, a);
        AF b1 = load_bfrag(sWd, NG, nt, 1, lane);
        a = WMMA_F16(ah1.v, b1.v, a);
        acc[g] = a;
      }
      float bi = sBd[16 * (0 * 4 + ht) + lanelo];
      float bf = sBd[16 * (1 * 4 + ht) + lanelo];
      float bg = sBd[16 * (2 * 4 + ht) + lanelo];
      float bo = sBd[16 * (3 * 4 + ht) + lanelo];
#pragma unroll
      for (int j = 0; j < 8; ++j) {
        float ig = fast_sigmoid(acc[0][j] + bi);
        float fg = fast_sigmoid(acc[1][j] + bf);
        float gg = fast_tanh(acc[2][j] + bg);
        float og = fast_sigmoid(acc[3][j] + bo);
        float cn = fg * c[ht][j] + ig * gg;
        c[ht][j] = cn;
        float hn = og * fast_tanh(cn);
        int m = j + (hi ? 8 : 0);
        hstage[m * HID + 16 * ht + lanelo] = (_Float16)hn;
      }
    }

    asm volatile("" ::: "memory");

    // projection: out_t = h @ Wp^T + bp
    AF ph0 = load_afrag_lds(harow, 0, hi);
    AF ph1 = load_afrag_lds(harow, 1, hi);
#pragma unroll
    for (int nt = 0; nt < 3; ++nt) {
      v8f a = {};
      AF b0 = load_bfrag(sWp, WPN, nt, 0, lane);
      a = WMMA_F16(ph0.v, b0.v, a);
      AF b1 = load_bfrag(sWp, WPN, nt, 1, lane);
      a = WMMA_F16(ph1.v, b1.v, a);
      int d = 16 * nt + lanelo;
      if (d < DIN) {
        float bb = sBp[d];
#pragma unroll
        for (int j = 0; j < 8; ++j) {
          int m = j + (hi ? 8 : 0);
          out[(size_t)(mBase + m) * SEQ * DIN + t * DIN + d] = a[j] + bb;
        }
      }
    }
  }
}

extern "C" void kernel_launch(void* const* d_in, const int* in_sizes, int n_in,
                              void* d_out, int out_size, void* d_ws, size_t ws_size,
                              hipStream_t stream) {
  (void)in_sizes; (void)n_in; (void)out_size; (void)d_ws; (void)ws_size;
  const float* x     = (const float*)d_in[0];
  const float* Wih_e = (const float*)d_in[1];
  const float* Whh_e = (const float*)d_in[2];
  const float* bih_e = (const float*)d_in[3];
  const float* bhh_e = (const float*)d_in[4];
  const float* Wih_d = (const float*)d_in[5];
  const float* Whh_d = (const float*)d_in[6];
  const float* bih_d = (const float*)d_in[7];
  const float* bhh_d = (const float*)d_in[8];
  const float* Wp    = (const float*)d_in[9];
  const float* bp    = (const float*)d_in[10];
  float* out = (float*)d_out;

  dim3 grid(BATCH / 128);
  dim3 block(256);
  lstm_ae_kernel<<<grid, block, 0, stream>>>(x, Wih_e, Whh_e, bih_e, bhh_e,
                                             Wih_d, Whh_d, bih_d, bhh_d,
                                             Wp, bp, out);
}
